// RVIT_1_91070486544616
// MI455X (gfx1250) — matrix operand order repaired and checked by reading, not compiled, not run on hardware
//
#include <hip/hip_runtime.h>
#include <hip/hip_bf16.h>
#include <math.h>

typedef __attribute__((ext_vector_type(16))) __bf16 v16bf;
typedef __attribute__((ext_vector_type(8)))  __bf16 v8bf;
typedef __attribute__((ext_vector_type(4)))  __bf16 v4bf;
typedef __attribute__((ext_vector_type(8)))  float  v8f;

#define LDF   132   // padded leading dim for 64x128 fp32 LDS tiles
#define LDSS  68    // padded leading dim for 64x64 score tile

// ---------------- WMMA core ----------------
__device__ __forceinline__ v8f wmma_bf16(v16bf a, v16bf b, v8f c) {
  // D = A(16x32) * B(32x16) + C(16x16), f32 accumulate
  return __builtin_amdgcn_wmma_f32_16x16x32_bf16(false, a, false, b, (short)0, c,
                                                 false, false);
}

// ---- A fragment: 16x32 tile of row-major X at (m0,k0).
// Per lane the 16 elements are p[0..15] (contiguous k, matches B loaders)
// with p = row(m0 + lane&15) + k0 + (lane>=16 ? 16 : 0).
__device__ __forceinline__ v16bf ldA(const float* X, int ld, int m0, int k0, int lane) {
  const float* p = X + (size_t)(m0 + (lane & 15)) * ld + k0 + (lane >> 4) * 16;
  const float4 q0 = *(const float4*)(p);
  const float4 q1 = *(const float4*)(p + 4);
  const float4 q2 = *(const float4*)(p + 8);
  const float4 q3 = *(const float4*)(p + 12);
  v16bf a;
  a[0] = (__bf16)q0.x;  a[1] = (__bf16)q0.y;  a[2] = (__bf16)q0.z;  a[3] = (__bf16)q0.w;
  a[4] = (__bf16)q1.x;  a[5] = (__bf16)q1.y;  a[6] = (__bf16)q1.z;  a[7] = (__bf16)q1.w;
  a[8] = (__bf16)q2.x;  a[9] = (__bf16)q2.y;  a[10] = (__bf16)q2.z; a[11] = (__bf16)q2.w;
  a[12] = (__bf16)q3.x; a[13] = (__bf16)q3.y; a[14] = (__bf16)q3.z; a[15] = (__bf16)q3.w;
  return a;
}
__device__ __forceinline__ v16bf ldA(const __bf16* X, int ld, int m0, int k0, int lane) {
  const __bf16* p = X + (size_t)(m0 + (lane & 15)) * ld + k0 + (lane >> 4) * 16;
  const v8bf lo = *(const v8bf*)(p);        // 16B load
  const v8bf hi = *(const v8bf*)(p + 8);    // 16B load
  v16bf a;
#pragma unroll
  for (int i = 0; i < 8; ++i) { a[i] = lo[i]; a[8 + i] = hi[i]; }
  return a;
}

// ---- B fragment with B[k][n] = W[n][k] (row-major W, rows = output features).
// Per lane: 16 contiguous elements p[0..15], p = row(n0 + lane&15) + k0 + (lane>=16?16:0)
__device__ __forceinline__ v16bf ldBT(const float* W, int ld, int n0, int k0, int lane) {
  const float* p = W + (size_t)(n0 + (lane & 15)) * ld + k0 + (lane >> 4) * 16;
  const float4 q0 = *(const float4*)(p);
  const float4 q1 = *(const float4*)(p + 4);
  const float4 q2 = *(const float4*)(p + 8);
  const float4 q3 = *(const float4*)(p + 12);
  v16bf b;
  b[0] = (__bf16)q0.x;  b[1] = (__bf16)q0.y;  b[2] = (__bf16)q0.z;  b[3] = (__bf16)q0.w;
  b[4] = (__bf16)q1.x;  b[5] = (__bf16)q1.y;  b[6] = (__bf16)q1.z;  b[7] = (__bf16)q1.w;
  b[8] = (__bf16)q2.x;  b[9] = (__bf16)q2.y;  b[10] = (__bf16)q2.z; b[11] = (__bf16)q2.w;
  b[12] = (__bf16)q3.x; b[13] = (__bf16)q3.y; b[14] = (__bf16)q3.z; b[15] = (__bf16)q3.w;
  return b;
}
__device__ __forceinline__ v16bf ldBT(const __bf16* W, int ld, int n0, int k0, int lane) {
  const __bf16* p = W + (size_t)(n0 + (lane & 15)) * ld + k0 + (lane >> 4) * 16;
  const v8bf lo = *(const v8bf*)(p);
  const v8bf hi = *(const v8bf*)(p + 8);
  v16bf b;
#pragma unroll
  for (int i = 0; i < 8; ++i) { b[i] = lo[i]; b[8 + i] = hi[i]; }
  return b;
}

// ---- B fragment with B[k][n] = X[k][n] (row-major X, e.g. V matrix), bf16 source
__device__ __forceinline__ v16bf ldB(const __bf16* X, int ld, int k0, int n0, int lane) {
  const __bf16* p = X + (size_t)(k0 + (lane >> 4) * 16) * ld + n0 + (lane & 15);
  v16bf b;
#pragma unroll
  for (int v = 0; v < 8; ++v) {
    b[2 * v]     = p[(2 * v) * ld];
    b[2 * v + 1] = p[(2 * v + 1) * ld];
  }
  return b;
}

// Generic 64xFO GEMM. Wave owns ONE n-tile (column strip) and NM m-tiles held as
// register accumulators, so each B fragment is loaded once and reused NM times.
template <bool BT, int NM, typename TA, typename TB, typename Epi>
__device__ __forceinline__ void gemm64(const TA* A, int lda, const TB* Bm, int ldb,
                                       int FO, int K, int tid, Epi epi) {
  const int w = tid >> 5, lane = tid & 31;
  const int ntc = FO >> 4;
  const int n0 = (w % ntc) << 4;
  const int mb = (w / ntc) * NM;
  v8f acc[NM] = {};
  for (int k = 0; k < K; k += 32) {
    v16bf b;
    if constexpr (BT) b = ldBT(Bm, ldb, n0, k, lane);
    else              b = ldB(Bm, ldb, k, n0, lane);
#pragma unroll
    for (int i = 0; i < NM; ++i) {
      v16bf a = ldA(A, lda, (mb + i) << 4, k, lane);
      acc[i] = wmma_bf16(a, b, acc[i]);
    }
  }
#pragma unroll
  for (int i = 0; i < NM; ++i) epi((mb + i) << 4, n0, lane, acc[i]);
}

// ---------------- wave32 row ops ----------------
__device__ __forceinline__ float wred_sum(float x) {
#pragma unroll
  for (int o = 16; o > 0; o >>= 1) x += __shfl_xor(x, o, 32);
  return x;
}
__device__ __forceinline__ float wred_max(float x) {
#pragma unroll
  for (int o = 16; o > 0; o >>= 1) x = fmaxf(x, __shfl_xor(x, o, 32));
  return x;
}

// LayerNorm over 128 cols, 64 rows; wave w handles rows w, w+8, ...
// Optionally mirrors the normalized result into a bf16 buffer (stride 128).
__device__ __forceinline__ void ln_rows(float* buf, int ld, const float* s,
                                        const float* b, int tid, __bf16* mir) {
  const int w = tid >> 5, lane = tid & 31;
  for (int r = w; r < 64; r += 8) {
    float* row = buf + (size_t)r * ld;
    float x0 = row[lane], x1 = row[lane + 32], x2 = row[lane + 64], x3 = row[lane + 96];
    float m = wred_sum(x0 + x1 + x2 + x3) * (1.f / 128.f);
    float d0 = x0 - m, d1 = x1 - m, d2 = x2 - m, d3 = x3 - m;
    float v = wred_sum(d0 * d0 + d1 * d1 + d2 * d2 + d3 * d3) * (1.f / 128.f);
    float inv = rsqrtf(v + 1e-5f);
    float y0 = d0 * inv * s[lane]      + b[lane];
    float y1 = d1 * inv * s[lane + 32] + b[lane + 32];
    float y2 = d2 * inv * s[lane + 64] + b[lane + 64];
    float y3 = d3 * inv * s[lane + 96] + b[lane + 96];
    row[lane] = y0; row[lane + 32] = y1; row[lane + 64] = y2; row[lane + 96] = y3;
    if (mir) {
      __bf16* mrow = mir + (size_t)r * 128;
      mrow[lane] = (__bf16)y0;      mrow[lane + 32] = (__bf16)y1;
      mrow[lane + 64] = (__bf16)y2; mrow[lane + 96] = (__bf16)y3;
    }
  }
}

// Softmax over 64 cols, 64 rows.
__device__ __forceinline__ void softmax_rows(float* S, int ld, int tid) {
  const int w = tid >> 5, lane = tid & 31;
  for (int r = w; r < 64; r += 8) {
    float* row = S + (size_t)r * ld;
    float x0 = row[lane], x1 = row[lane + 32];
    float mx = wred_max(fmaxf(x0, x1));
    float e0 = __expf(x0 - mx), e1 = __expf(x1 - mx);
    float inv = 1.f / wred_sum(e0 + e1);
    row[lane] = e0 * inv;
    row[lane + 32] = e1 * inv;
  }
}

__device__ __forceinline__ float gelu_erf(float x) {
  return 0.5f * x * (1.f + erff(x * 0.70710678118654752f));
}

// Fused dual GEMM + sigmoid(alpha) mix -> bf16 staging for one head's Q/K/V.
template <typename WT>
__device__ __forceinline__ void qkv_head(__bf16* dst, const __bf16* curb,
                                         const __bf16* preb,
                                         const WT* Wc, const float* bc,
                                         const WT* Wp, const float* bp,
                                         const float* al, int tid) {
  const int w = tid >> 5, lane = tid & 31;
  const int n0 = w << 4;  // 8 waves <-> 8 n-tiles
  v8f ac[4] = {}, ap[4] = {};
  for (int k = 0; k < 128; k += 32) {
    v16bf bcf = ldBT(Wc, 128, n0, k, lane);
    v16bf bpf = ldBT(Wp, 128, n0, k, lane);
#pragma unroll
    for (int i = 0; i < 4; ++i) {
      v16bf a1 = ldA(curb, 128, i << 4, k, lane);
      ac[i] = wmma_bf16(a1, bcf, ac[i]);
      v16bf a2 = ldA(preb, 128, i << 4, k, lane);
      ap[i] = wmma_bf16(a2, bpf, ap[i]);
    }
  }
  const int g = lane >> 4, n = n0 + (lane & 15);
  const float aa = 1.f / (1.f + __expf(-al[n]));
  const float bcv = bc[n], bpv = bp[n];
#pragma unroll
  for (int i = 0; i < 4; ++i) {
    const int m0 = i << 4;
#pragma unroll
    for (int r = 0; r < 8; ++r) {
      float v = aa * (ac[i][r] + bcv) + (1.f - aa) * (ap[i][r] + bpv);
      dst[(size_t)(m0 + r + 8 * g) * 128 + n] = (__bf16)v;
    }
  }
}

// ---------------- weight pre-conversion (fp32 -> bf16 in d_ws) ----------------
__global__ __launch_bounds__(256) void cvt_f32_to_bf16(const float* __restrict__ src,
                                                       __bf16* __restrict__ dst,
                                                       int n4) {
  int i = blockIdx.x * 256 + threadIdx.x;
  if (i < n4) {
    float4 q = ((const float4*)src)[i];
    v4bf o;
    o[0] = (__bf16)q.x; o[1] = (__bf16)q.y; o[2] = (__bf16)q.z; o[3] = (__bf16)q.w;
    ((v4bf*)dst)[i] = o;
  }
}

// ---------------- fused RViT block kernel ----------------
// One workgroup per (b,n) slice; 64x128 activation tile lives in LDS across layers.
// WT = weight element type (__bf16 when pre-converted into d_ws, float fallback).
template <typename WT>
__global__ __launch_bounds__(256, 1) void rvit_fused_kernel(
    const float* __restrict__ seq, const float* __restrict__ Hpre,
    const WT* __restrict__ pro1_w, const float* __restrict__ pro1_b,
    const WT* __restrict__ pro2_w, const float* __restrict__ pro2_b,
    const float* __restrict__ ln_cur_s, const float* __restrict__ ln_cur_b,
    const float* __restrict__ ln_pre_s, const float* __restrict__ ln_pre_b,
    const WT* __restrict__ cur_qkv_w, const float* __restrict__ cur_qkv_b,
    const WT* __restrict__ pre_qkv_w, const float* __restrict__ pre_qkv_b,
    const float* __restrict__ alpha,
    const WT* __restrict__ proj_w, const float* __restrict__ proj_b,
    const float* __restrict__ bn1_s, const float* __restrict__ bn1_b,
    const float* __restrict__ bn2_s, const float* __restrict__ bn2_b,
    const float* __restrict__ ln_o_s, const float* __restrict__ ln_o_b,
    const WT* __restrict__ ffn_w, const float* __restrict__ ffn_b,
    float* __restrict__ out) {
  constexpr int P = 64, D = 128, F = 128, HH = 8, L = 4, HF = 1024;
  constexpr float SCALE = 0.08838834764831845f;  // 128^-0.5

  extern __shared__ char smem[];
  float*  s_cur  = (float*)smem;               // 64x132 f32 (persistent activation)
  float*  s_pre  = s_cur + P * LDF;            // 64x132 f32 (pre act / ffn out)
  float*  s_acc  = s_pre + P * LDF;            // 64x132 f32 (proj accumulator)
  float*  s_tmp  = s_acc + P * LDF;            // 64x132 f32 (A_h / O_hat)
  float*  s_S    = s_tmp + P * LDF;            // 64x68  f32 (scores)
  __bf16* s_Q    = (__bf16*)(s_S + P * LDSS);  // 64x128 bf16 (Q / A_h / O_hat bf16)
  __bf16* s_K    = s_Q + P * F;                // 64x128 bf16
  __bf16* s_V    = s_K + P * F;                // 64x128 bf16
  __bf16* s_curb = s_V + P * F;                // 64x128 bf16 mirror of cur
  __bf16* s_preb = s_curb + P * F;             // 64x128 bf16 mirror of pre

  const int tid = threadIdx.x;
  const int wg  = blockIdx.x;  // b*N + n
  const float* seq_t = seq + (size_t)wg * P * D;

  for (int l = 0; l < L; ++l) {
    // ---- cur = LN(seq @ pro1^T + b) (layer 0 only) ----
    if (l == 0) {
      gemm64<true, 4>(seq_t, D, pro1_w, D, 128, 128, tid,
                      [&](int m0, int n0, int lane, v8f acc) {
                        int g = lane >> 4, n = n0 + (lane & 15);
                        float bv = pro1_b[n];
#pragma unroll
                        for (int r = 0; r < 8; ++r)
                          s_cur[(size_t)(m0 + r + 8 * g) * LDF + n] = acc[r] + bv;
                      });
      __syncthreads();
      ln_rows(s_cur, LDF, ln_cur_s, ln_cur_b, tid, s_curb);
      __syncthreads();
    }

    // ---- pre = LN(H_pre[l] @ pro2^T + b) ----
    const float* Hp  = Hpre + ((size_t)l * 128 + wg) * P * D;
    const WT*    W2  = pro2_w + (size_t)l * F * D;
    const float* b2v = pro2_b + l * F;
    gemm64<true, 4>(Hp, D, W2, D, 128, 128, tid,
                    [&](int m0, int n0, int lane, v8f acc) {
                      int g = lane >> 4, n = n0 + (lane & 15);
                      float bv = b2v[n];
#pragma unroll
                      for (int r = 0; r < 8; ++r)
                        s_pre[(size_t)(m0 + r + 8 * g) * LDF + n] = acc[r] + bv;
                    });
    __syncthreads();
    ln_rows(s_pre, LDF, ln_pre_s + l * F, ln_pre_b + l * F, tid, s_preb);

    // ---- init proj accumulator with proj bias ----
    const float* pb = proj_b + l * F;
    for (int i = tid; i < P * F; i += 256)
      s_acc[(size_t)(i >> 7) * LDF + (i & 127)] = pb[i & 127];
    __syncthreads();

    const WT*    Wc0 = cur_qkv_w + (size_t)l * 3 * HF * F;
    const WT*    Wp0 = pre_qkv_w + (size_t)l * 3 * HF * F;
    const float* bc0 = cur_qkv_b + (size_t)l * 3 * HF;
    const float* bp0 = pre_qkv_b + (size_t)l * 3 * HF;
    const float* al0 = alpha + (size_t)l * 3 * HF;
    const WT*    Wpr = proj_w + (size_t)l * F * HF;

    for (int h = 0; h < HH; ++h) {
      // Q/K/V with sigmoid(alpha) mixing of cur- and pre-paths
#pragma unroll
      for (int q = 0; q < 3; ++q) {
        __bf16* dst = (q == 0) ? s_Q : (q == 1) ? s_K : s_V;
        size_t ro = (size_t)(q * HF + h * F);
        qkv_head(dst, s_curb, s_preb, Wc0 + ro * F, bc0 + ro, Wp0 + ro * F,
                 bp0 + ro, al0 + q * HF + h * F, tid);
      }
      if (h + 1 < HH)  // warm caches for next head's weight slices
        __builtin_prefetch(Wc0 + (size_t)(h + 1) * F * F, 0, 1);
      __syncthreads();

      // S = Q @ K^T * scale
      gemm64<true, 2>(s_Q, F, s_K, F, 64, 128, tid,
                      [&](int m0, int n0, int lane, v8f acc) {
                        int g = lane >> 4, n = n0 + (lane & 15);
#pragma unroll
                        for (int r = 0; r < 8; ++r)
                          s_S[(size_t)(m0 + r + 8 * g) * LDSS + n] = acc[r] * SCALE;
                      });
      __syncthreads();
      softmax_rows(s_S, LDSS, tid);
      __syncthreads();

      // A_h = softmax(S) @ V
      gemm64<false, 4>(s_S, LDSS, s_V, F, 128, 64, tid,
                       [&](int m0, int n0, int lane, v8f acc) {
                         int g = lane >> 4, n = n0 + (lane & 15);
#pragma unroll
                         for (int r = 0; r < 8; ++r)
                           s_tmp[(size_t)(m0 + r + 8 * g) * LDF + n] = acc[r];
                       });
      __syncthreads();
      // bn1 LN; mirror bf16 result into s_Q (free now) for the proj GEMM
      ln_rows(s_tmp, LDF, bn1_s + l * F, bn1_b + l * F, tid, s_Q);
      __syncthreads();

      // proj accumulation: acc += A_h @ proj_w[:, h*F:(h+1)*F]^T
      gemm64<true, 4>(s_Q, 128, Wpr + h * F, HF, 128, 128, tid,
                      [&](int m0, int n0, int lane, v8f acc) {
                        int g = lane >> 4, n = n0 + (lane & 15);
#pragma unroll
                        for (int r = 0; r < 8; ++r)
                          s_acc[(size_t)(m0 + r + 8 * g) * LDF + n] += acc[r];
                      });
      __syncthreads();
    }

    // ---- epilogue: bn2, residuals, ln_o, FFN ----
    ln_rows(s_acc, LDF, bn2_s + l * F, bn2_b + l * F, tid, nullptr);  // A = LN(proj)
    __syncthreads();
    for (int i = tid; i < P * F; i += 256) {
      size_t idx = (size_t)(i >> 7) * LDF + (i & 127);
      float cv = s_cur[idx];
      s_tmp[idx] = cv + (s_acc[idx] + gelu_erf(cv));  // cur + (A + gelu(cur))
    }
    __syncthreads();
    // O_hat = LN(...); mirror bf16 into s_Q for the FFN GEMM
    ln_rows(s_tmp, LDF, ln_o_s + l * F, ln_o_b + l * F, tid, s_Q);
    __syncthreads();

    const WT*    Wf  = ffn_w + (size_t)l * F * F;
    const float* bfv = ffn_b + l * F;
    gemm64<true, 4>(s_Q, 128, Wf, F, 128, 128, tid,
                    [&](int m0, int n0, int lane, v8f acc) {
                      int g = lane >> 4, n = n0 + (lane & 15);
                      float bv = bfv[n];
#pragma unroll
                      for (int r = 0; r < 8; ++r)
                        s_pre[(size_t)(m0 + r + 8 * g) * LDF + n] = acc[r] + bv;
                    });
    __syncthreads();
    for (int i = tid; i < P * F; i += 256) {
      size_t idx = (size_t)(i >> 7) * LDF + (i & 127);
      float nv = s_pre[idx] + gelu_erf(s_tmp[idx]);  // next cur
      s_cur[idx] = nv;
      s_curb[i]  = (__bf16)nv;  // refresh bf16 mirror for next layer's QKV
    }
    __syncthreads();
  }

  // ---- write result (B,N,P,F) ----
  for (int i = tid; i < P * F; i += 256)
    out[(size_t)wg * P * F + i] = s_cur[(size_t)(i >> 7) * LDF + (i & 127)];
}

extern "C" void kernel_launch(void* const* d_in, const int* in_sizes, int n_in,
                              void* d_out, int out_size, void* d_ws, size_t ws_size,
                              hipStream_t stream) {
  (void)in_sizes; (void)n_in; (void)out_size;
  const float* seq       = (const float*)d_in[0];
  const float* Hpre      = (const float*)d_in[1];
  const float* pro1_w    = (const float*)d_in[2];
  const float* pro1_b    = (const float*)d_in[3];
  const float* pro2_w    = (const float*)d_in[4];
  const float* pro2_b    = (const float*)d_in[5];
  const float* ln_cur_s  = (const float*)d_in[6];
  const float* ln_cur_b  = (const float*)d_in[7];
  const float* ln_pre_s  = (const float*)d_in[8];
  const float* ln_pre_b  = (const float*)d_in[9];
  const float* cur_qkv_w = (const float*)d_in[10];
  const float* cur_qkv_b = (const float*)d_in[11];
  const float* pre_qkv_w = (const float*)d_in[12];
  const float* pre_qkv_b = (const float*)d_in[13];
  const float* alpha     = (const float*)d_in[14];
  const float* proj_w    = (const float*)d_in[15];
  const float* proj_b    = (const float*)d_in[16];
  const float* bn1_s     = (const float*)d_in[17];
  const float* bn1_b     = (const float*)d_in[18];
  const float* bn2_s     = (const float*)d_in[19];
  const float* bn2_b     = (const float*)d_in[20];
  const float* ln_o_s    = (const float*)d_in[21];
  const float* ln_o_b    = (const float*)d_in[22];
  const float* ffn_w     = (const float*)d_in[23];
  const float* ffn_b     = (const float*)d_in[24];
  float* out = (float*)d_out;

  // LDS: 4x(64x132 f32) + 64x68 f32 + 5x(64x128 bf16) = 234,496 bytes (< 320KB/WG)
  const size_t lds_bytes =
      (size_t)(4 * 64 * LDF + 64 * LDSS) * sizeof(float) + (size_t)5 * 64 * 128 * 2;

  // Weight element counts (all fp32 in d_in)
  const size_t N_PRO1 = 4ull * 128 * 128;        //   65,536
  const size_t N_PRO2 = 4ull * 128 * 128;        //   65,536
  const size_t N_QKV  = 4ull * 3072 * 128;       // 1,572,864 (each of cur/pre)
  const size_t N_PROJ = 4ull * 128 * 1024;       //   524,288
  const size_t N_FFN  = 4ull * 128 * 128;        //   65,536
  const size_t N_W    = N_PRO1 + N_PRO2 + 2 * N_QKV + N_PROJ + N_FFN;  // 3,866,624

  if (ws_size >= N_W * 2) {
    // Pre-convert all weight matrices fp32 -> bf16 into d_ws (one cheap pass,
    // ~8 MB; removes all weight v_cvt chains from the fused kernel's hot loop).
    __bf16* wsb = (__bf16*)d_ws;
    __bf16* w_pro1 = wsb;
    __bf16* w_pro2 = w_pro1 + N_PRO1;
    __bf16* w_cqkv = w_pro2 + N_PRO2;
    __bf16* w_pqkv = w_cqkv + N_QKV;
    __bf16* w_proj = w_pqkv + N_QKV;
    __bf16* w_ffn  = w_proj + N_PROJ;

    auto cvt = [&](const float* s, __bf16* d, size_t n) {
      int n4 = (int)(n / 4);
      cvt_f32_to_bf16<<<dim3((n4 + 255) / 256), dim3(256), 0, stream>>>(s, d, n4);
    };
    cvt(pro1_w, w_pro1, N_PRO1);
    cvt(pro2_w, w_pro2, N_PRO2);
    cvt(cur_qkv_w, w_cqkv, N_QKV);
    cvt(pre_qkv_w, w_pqkv, N_QKV);
    cvt(proj_w, w_proj, N_PROJ);
    cvt(ffn_w, w_ffn, N_FFN);

    rvit_fused_kernel<__bf16><<<dim3(128), dim3(256), lds_bytes, stream>>>(
        seq, Hpre, w_pro1, pro1_b, w_pro2, pro2_b, ln_cur_s, ln_cur_b, ln_pre_s,
        ln_pre_b, w_cqkv, cur_qkv_b, w_pqkv, pre_qkv_b, alpha, w_proj, proj_b,
        bn1_s, bn1_b, bn2_s, bn2_b, ln_o_s, ln_o_b, w_ffn, ffn_b, out);
  } else {
    // Fallback: read fp32 weights directly (convert in-kernel).
    rvit_fused_kernel<float><<<dim3(128), dim3(256), lds_bytes, stream>>>(
        seq, Hpre, pro1_w, pro1_b, pro2_w, pro2_b, ln_cur_s, ln_cur_b, ln_pre_s,
        ln_pre_b, cur_qkv_w, cur_qkv_b, pre_qkv_w, pre_qkv_b, alpha, proj_w, proj_b,
        bn1_s, bn1_b, bn2_s, bn2_b, ln_o_s, ln_o_b, ffn_w, ffn_b, out);
  }
}
